// AttentionLayer_80625126081332
// MI455X (gfx1250) — matrix operand order repaired and checked by reading, not compiled, not run on hardware
//
#include <hip/hip_runtime.h>

#define SLEN 4096
#define DDIM 1024
#define BATCH 4

typedef __attribute__((ext_vector_type(16))) __bf16 v16bf;
typedef __attribute__((ext_vector_type(8)))  __bf16 v8bf;
typedef __attribute__((ext_vector_type(8)))  float  v8f;

static __device__ inline v16bf cat8(v8bf lo, v8bf hi) {
  return __builtin_shufflevector(lo, hi, 0,1,2,3,4,5,6,7,8,9,10,11,12,13,14,15);
}
static __device__ inline __bf16 bf(float x) { return (__bf16)x; }

// gfx1250 async global->LDS copy (ASYNCcnt-tracked).  lds = LDS byte offset
// (low 32 bits of the generic shared pointer), g = 64-bit global address.
static __device__ inline void async_ld128(unsigned lds, const void* g) {
  asm volatile("global_load_async_to_lds_b128 %0, %1, off"
               :: "v"(lds), "v"(g) : "memory");
}
#define WAIT_ASYNC(n) asm volatile("s_wait_asynccnt " #n ::: "memory")

static __device__ inline unsigned lds_off(const void* p) {
  return (unsigned)(size_t)p;   // LDS generic addr: low 32 bits = LDS offset
}

// ---------------------------------------------------------------------------
// Kernel 0: fp32 -> bf16 conversion (x and the three weight matrices).
// ---------------------------------------------------------------------------
__global__ __launch_bounds__(256)
void to_bf16(const float* __restrict__ src, __bf16* __restrict__ dst, int n8) {
  const int i = blockIdx.x * 256 + threadIdx.x;   // 8-element chunk index
  if (i >= n8) return;
  const float4 f0 = *(const float4*)(src + (size_t)i * 8);
  const float4 f1 = *(const float4*)(src + (size_t)i * 8 + 4);
  v8bf p;
  p[0] = bf(f0.x); p[1] = bf(f0.y); p[2] = bf(f0.z); p[3] = bf(f0.w);
  p[4] = bf(f1.x); p[5] = bf(f1.y); p[6] = bf(f1.z); p[7] = bf(f1.w);
  *(v8bf*)(dst + (size_t)i * 8) = p;
}

// ---------------------------------------------------------------------------
// Kernel 1: QKV projections, bf16 inputs, async double-buffered LDS staging.
// y = x @ W^T + b.  Q,K stored row-major bf16 [B*S, D]; V stored transposed
// bf16 [D, B*S].  blockIdx: x = M-tile(128), y = N-tile(128), z = 0/1/2.
// ---------------------------------------------------------------------------
__global__ __launch_bounds__(256)
void qkv_gemm(const __bf16* __restrict__ xb,
              const __bf16* __restrict__ Wqb, const float* __restrict__ bq,
              const __bf16* __restrict__ Wkb, const float* __restrict__ bk,
              const __bf16* __restrict__ Wvb, const float* __restrict__ bv,
              __bf16* __restrict__ Qb, __bf16* __restrict__ Kb,
              __bf16* __restrict__ Vt)
{
  __shared__ __bf16 As[2][128][32];   // x tile  [m][k], double buffered
  __shared__ __bf16 Bs[2][128][32];   // W tile  [n][k], double buffered

  const int tid  = threadIdx.x;
  const int wid  = tid >> 5;
  const int lane = tid & 31;
  const int ln16 = lane & 15;
  const int hi   = lane >> 4;

  const int m0 = blockIdx.x * 128;
  const int n0 = blockIdx.y * 128;
  const int z  = blockIdx.z;

  const __bf16* W  = (z == 0) ? Wqb : (z == 1) ? Wkb : Wvb;
  const float*  bb = (z == 0) ? bq  : (z == 1) ? bk  : bv;

  const int wr = wid & 3;   // 32-row group
  const int wc = wid >> 2;  // 64-col group

  const __bf16* xg = xb + (size_t)m0 * DDIM;
  const __bf16* wg = W  + (size_t)n0 * DDIM;

  const unsigned aL[2] = { lds_off(&As[0][0][0]), lds_off(&As[1][0][0]) };
  const unsigned bL[2] = { lds_off(&Bs[0][0][0]), lds_off(&Bs[1][0][0]) };

  // per-thread staging assignment: 4 async b128 per tile pair
  const int sc = tid & 3;        // 8-bf16 chunk within the 32-wide k slab
  const int sr = tid >> 2;       // 0..63 (rows sr and sr+64)

  // issue helper (macro to keep immediates simple)
#define STAGE(buf, kkv)                                                        \
  do {                                                                         \
    const __bf16* xa = xg + (size_t)sr * DDIM + (kkv) + sc * 8;                \
    const __bf16* wa = wg + (size_t)sr * DDIM + (kkv) + sc * 8;                \
    async_ld128(aL[buf] + (sr * 32 + sc * 8) * 2, xa);                         \
    async_ld128(aL[buf] + ((sr + 64) * 32 + sc * 8) * 2, xa + 64 * DDIM);      \
    async_ld128(bL[buf] + (sr * 32 + sc * 8) * 2, wa);                         \
    async_ld128(bL[buf] + ((sr + 64) * 32 + sc * 8) * 2, wa + 64 * DDIM);      \
  } while (0)

  v8f acc[2][4] = {};

  STAGE(0, 0);
  for (int it = 0; it < DDIM / 32; ++it) {
    const int buf = it & 1;
    if (it + 1 < DDIM / 32) {
      STAGE(buf ^ 1, (it + 1) * 32);
      WAIT_ASYNC(0x4);           // current slab done; next slab in flight
    } else {
      WAIT_ASYNC(0x0);
    }
    __syncthreads();

    v16bf afrag[2];
#pragma unroll
    for (int mt = 0; mt < 2; ++mt) {
      const int row = wr * 32 + mt * 16 + ln16;
      const int off = hi ? 8 : 0;
      afrag[mt] = cat8(*(const v8bf*)&As[buf][row][off],
                       *(const v8bf*)&As[buf][row][off + 16]);
    }
#pragma unroll
    for (int nt = 0; nt < 4; ++nt) {
      const int row = wc * 64 + nt * 16 + ln16;
      const int off = hi ? 8 : 0;
      v16bf bfrag = cat8(*(const v8bf*)&Bs[buf][row][off],
                         *(const v8bf*)&Bs[buf][row][off + 16]);
#pragma unroll
      for (int mt = 0; mt < 2; ++mt)
        acc[mt][nt] = __builtin_amdgcn_wmma_f32_16x16x32_bf16(
            false, afrag[mt], false, bfrag, (short)0, acc[mt][nt], false, false);
    }
    __syncthreads();             // done reading buf before it is re-staged
  }
#undef STAGE

  // epilogue: bias add, bf16 convert, store
#pragma unroll
  for (int mt = 0; mt < 2; ++mt) {
#pragma unroll
    for (int nt = 0; nt < 4; ++nt) {
      const int ncol = n0 + wc * 64 + nt * 16 + ln16;
      const float bias = bb[ncol];
      const int mrow0 = m0 + wr * 32 + mt * 16 + hi * 8;
      if (z < 2) {
        __bf16* dst = (z == 0) ? Qb : Kb;
#pragma unroll
        for (int r = 0; r < 8; ++r)
          dst[(size_t)(mrow0 + r) * DDIM + ncol] = bf(acc[mt][nt][r] + bias);
      } else {
        v8bf pk;
#pragma unroll
        for (int r = 0; r < 8; ++r) pk[r] = bf(acc[mt][nt][r] + bias);
        *(v8bf*)&Vt[(size_t)ncol * (BATCH * SLEN) + mrow0] = pk;
      }
    }
  }
}

// ---------------------------------------------------------------------------
// Kernel 2: flash attention.  blockIdx.x = 16-query tile, blockIdx.y = batch.
// 8 waves: scores -> wave w handles keys [w*16, w*16+16) of the 128-key tile;
// PV -> wave w accumulates output d-columns [w*128, w*128+128).
// ---------------------------------------------------------------------------
__global__ __launch_bounds__(256)
void flash_attn(const __bf16* __restrict__ Qb, const __bf16* __restrict__ Kb,
                const __bf16* __restrict__ Vt, float* __restrict__ out)
{
  __shared__ __bf16 Qs[16][1024];   // 32 KB query tile
  __shared__ float  Ssc[16][128];
  __shared__ __bf16 Pt[16][128];
  __shared__ float  red[16][16];
  __shared__ float  mst[16], lst[16], alp[16];

  const int tid  = threadIdx.x;
  const int wid  = tid >> 5;
  const int lane = tid & 31;
  const int ln16 = lane & 15;
  const int hi   = lane >> 4;

  const int b  = blockIdx.y;
  const int q0 = blockIdx.x * 16;

  const __bf16* Qg = Qb + ((size_t)b * SLEN + q0) * DDIM;
  const __bf16* Kg = Kb + (size_t)b * SLEN * DDIM;
  const __bf16* Vg = Vt + (size_t)b * SLEN;   // row stride BATCH*SLEN

  // stage Q tile via async global->LDS (2048 x 16B)
  {
    const unsigned q_lds = lds_off(&Qs[0][0]);
    for (int c = tid; c < 2048; c += 256)
      async_ld128(q_lds + c * 16, Qg + c * 8);
    WAIT_ASYNC(0x0);
  }
  if (tid < 16) { mst[tid] = -1e30f; lst[tid] = 0.f; }
  __syncthreads();

  v8f oacc[8] = {};

  for (int kt = 0; kt < SLEN / 128; ++kt) {
    const int key0 = kt * 128;

    // prefetch next key tile's K row for this lane
    if (kt + 1 < SLEN / 128)
      __builtin_prefetch(Kg + (size_t)(key0 + 128 + wid * 16 + ln16) * DDIM, 0, 1);

    // ---- scores: S[16 x 16] per wave, K-loop over D ----
    v8f sacc = {};
    const __bf16* krow = Kg + (size_t)(key0 + wid * 16 + ln16) * DDIM;
#pragma unroll 4
    for (int kk = 0; kk < 32; ++kk) {
      const int aoff = kk * 32 + hi * 8;
      v16bf afrag = cat8(*(const v8bf*)&Qs[ln16][aoff],
                         *(const v8bf*)&Qs[ln16][aoff + 16]);
      const int boff = kk * 32 + hi * 8;
      v16bf bfrag = cat8(*(const v8bf*)(krow + boff),
                         *(const v8bf*)(krow + boff + 16));
      sacc = __builtin_amdgcn_wmma_f32_16x16x32_bf16(
          false, afrag, false, bfrag, (short)0, sacc, false, false);
    }
#pragma unroll
    for (int r = 0; r < 8; ++r)
      Ssc[r + hi * 8][wid * 16 + ln16] = sacc[r] * 0.03125f;  // 1/sqrt(1024)
    __syncthreads();

    // ---- online softmax over the 128-key tile ----
    const int row = tid >> 4, seg = tid & 15;
    {
      float lm = -1e30f;
#pragma unroll
      for (int c2 = 0; c2 < 8; ++c2) lm = fmaxf(lm, Ssc[row][seg * 8 + c2]);
      red[row][seg] = lm;
    }
    __syncthreads();
    if (seg == 0) {
      float m_new = mst[row];
#pragma unroll
      for (int s2 = 0; s2 < 16; ++s2) m_new = fmaxf(m_new, red[row][s2]);
      alp[row] = __expf(mst[row] - m_new);
      mst[row] = m_new;
    }
    __syncthreads();
    {
      const float m_new = mst[row];
      float psum = 0.f;
#pragma unroll
      for (int c2 = 0; c2 < 8; ++c2) {
        const float p = __expf(Ssc[row][seg * 8 + c2] - m_new);
        Pt[row][seg * 8 + c2] = bf(p);
        psum += p;
      }
      red[row][seg] = psum;
    }
    __syncthreads();
    if (seg == 0) {
      float ls = 0.f;
#pragma unroll
      for (int s2 = 0; s2 < 16; ++s2) ls += red[row][s2];
      lst[row] = lst[row] * alp[row] + ls;
    }
    __syncthreads();

    // ---- rescale O accumulators ----
    float fac[8];
#pragma unroll
    for (int r = 0; r < 8; ++r) fac[r] = alp[hi * 8 + r];
#pragma unroll
    for (int t = 0; t < 8; ++t)
#pragma unroll
      for (int r = 0; r < 8; ++r) oacc[t][r] *= fac[r];

    // ---- O += P @ V  (this wave's 128-wide d slice) ----
#pragma unroll
    for (int k2 = 0; k2 < 4; ++k2) {
      const int aoff = k2 * 32 + hi * 8;
      v16bf afrag = cat8(*(const v8bf*)&Pt[ln16][aoff],
                         *(const v8bf*)&Pt[ln16][aoff + 16]);
      const int kbase = key0 + k2 * 32 + hi * 8;
#pragma unroll
      for (int t = 0; t < 8; ++t) {
        const int dcol = wid * 128 + t * 16 + ln16;
        const __bf16* vrow = Vg + (size_t)dcol * (BATCH * SLEN) + kbase;
        v16bf bfrag = cat8(*(const v8bf*)vrow, *(const v8bf*)(vrow + 16));
        oacc[t] = __builtin_amdgcn_wmma_f32_16x16x32_bf16(
            false, afrag, false, bfrag, (short)0, oacc[t], false, false);
      }
    }
    __syncthreads();
  }

  // ---- final normalization + store ----
#pragma unroll
  for (int r = 0; r < 8; ++r) {
    const int row = r + hi * 8;
    const float inv = 1.f / lst[row];
#pragma unroll
    for (int t = 0; t < 8; ++t)
      out[((size_t)b * SLEN + q0 + row) * DDIM + wid * 128 + t * 16 + ln16] =
          oacc[t][r] * inv;
  }
}

// ---------------------------------------------------------------------------
extern "C" void kernel_launch(void* const* d_in, const int* in_sizes, int n_in,
                              void* d_out, int out_size, void* d_ws, size_t ws_size,
                              hipStream_t stream) {
  (void)in_sizes; (void)n_in; (void)out_size; (void)ws_size;
  const float* x  = (const float*)d_in[0];
  const float* Wq = (const float*)d_in[1];
  const float* bq = (const float*)d_in[2];
  const float* Wk = (const float*)d_in[3];
  const float* bk = (const float*)d_in[4];
  const float* Wv = (const float*)d_in[5];
  const float* bv = (const float*)d_in[6];
  float* out = (float*)d_out;

  const size_t NX = (size_t)BATCH * SLEN * DDIM;   // 16.7M
  const size_t NW = (size_t)DDIM * DDIM;           // 1M

  __bf16* Qb  = (__bf16*)d_ws;          // [B*S, D]
  __bf16* Kb  = Qb + NX;                // [B*S, D]
  __bf16* Vt  = Kb + NX;                // [D, B*S] (transposed)
  __bf16* xbb = Vt + NX;                // [B*S, D]
  __bf16* Wqb = xbb + NX;
  __bf16* Wkb = Wqb + NW;
  __bf16* Wvb = Wkb + NW;

  to_bf16<<<dim3((unsigned)(NX / 8 / 256)), 256, 0, stream>>>(x,  xbb, (int)(NX / 8));
  to_bf16<<<dim3((unsigned)(NW / 8 / 256)), 256, 0, stream>>>(Wq, Wqb, (int)(NW / 8));
  to_bf16<<<dim3((unsigned)(NW / 8 / 256)), 256, 0, stream>>>(Wk, Wkb, (int)(NW / 8));
  to_bf16<<<dim3((unsigned)(NW / 8 / 256)), 256, 0, stream>>>(Wv, Wvb, (int)(NW / 8));

  dim3 g1(BATCH * SLEN / 128, DDIM / 128, 3);
  qkv_gemm<<<g1, 256, 0, stream>>>(xbb, Wqb, bq, Wkb, bk, Wvb, bv, Qb, Kb, Vt);

  dim3 g2(SLEN / 16, BATCH);
  flash_attn<<<g2, 256, 0, stream>>>(Qb, Kb, Vt, out);
}
